// TrajectoryConsistencyLoss_52639119180385
// MI455X (gfx1250) — matrix operand order repaired and checked by reading, not compiled, not run on hardware
//
#include <hip/hip_runtime.h>
#include <math.h>

typedef _Float16 v16h __attribute__((ext_vector_type(16)));
typedef _Float16 v8h  __attribute__((ext_vector_type(8)));
typedef float    v8f  __attribute__((ext_vector_type(8)));

#define K_TOK 2048
#define BATCH 16
#define DIM   2048
#define EPSV  1e-8f
#define MAXN  (1.0f - 1e-8f)
#define RADIUS_T 2.0f
#define NITER 50
#define NACC  192   // sA[64] | gg[64] | pg[64]

__device__ __forceinline__ float wave_sum32(float v) {
    #pragma unroll
    for (int m = 16; m > 0; m >>= 1) v += __shfl_xor(v, m, 32);
    return v;
}

// ---------------------------------------------------------------------------
// Zero per-call state: center p, accumulator slots, output scalar.
__global__ void zero_state_kernel(float* __restrict__ p, float* __restrict__ acc,
                                  float* __restrict__ out) {
    int t = blockIdx.x * blockDim.x + threadIdx.x;
    if (t < DIM)  p[t]   = 0.0f;
    if (t < NACC) acc[t] = 0.0f;
    if (t == 0)   out[0] = 0.0f;
}

// ---------------------------------------------------------------------------
// tokens[k] = mean_b x[k,b,:] -> f16 row-major Xh (coalesced),
// x_sq[k] = |token_k|^2, w0h[k] = (arctanh(min(n,1-eps))/(n+eps))/K
__global__ void tokens_kernel(const float* __restrict__ in,
                              _Float16* __restrict__ Xh,
                              float* __restrict__ x_sq, _Float16* __restrict__ w0h) {
    const int k = blockIdx.x;
    const int t = threadIdx.x;
    __shared__ float red[256];
    const float* rowbase = in + (size_t)k * BATCH * DIM;
    float ss = 0.0f;
    #pragma unroll
    for (int j = 0; j < DIM / 256; ++j) {
        int d = t + j * 256;
        __builtin_prefetch(rowbase + d + 8 * DIM, 0, 1);   // global_prefetch_b8
        float s = 0.0f;
        #pragma unroll
        for (int b = 0; b < BATCH; ++b) s += rowbase[(size_t)b * DIM + d];
        float tok = s * (1.0f / BATCH);
        ss += tok * tok;
        Xh[(size_t)k * DIM + d] = (_Float16)tok;
    }
    red[t] = ss; __syncthreads();
    for (int s = 128; s > 0; s >>= 1) { if (t < s) red[t] += red[t + s]; __syncthreads(); }
    if (t == 0) {
        float xs = red[0];
        x_sq[k] = xs;
        float n = sqrtf(xs);
        float w0 = 0.0f;
        if (n >= EPSV) {
            float s  = fminf(n, MAXN);
            float at = 0.5f * logf((1.0f + s) / (1.0f - s));   // arctanh
            w0 = at / (n + EPSV) * (1.0f / (float)K_TOK);
        }
        w0h[k] = (_Float16)w0;
    }
}

// ---------------------------------------------------------------------------
// Coalesced 64x64 LDS-tiled transpose: Xth[d][k] = Xh[k][d].
__global__ void transpose_kernel(const _Float16* __restrict__ Xh,
                                 _Float16* __restrict__ Xth) {
    __shared__ _Float16 tile[64][72];          // +8 pad: no bank conflicts
    const int bk = blockIdx.x & 31;            // k-tile
    const int bd = blockIdx.x >> 5;            // d-tile
    const int k0 = bk * 64, d0 = bd * 64;
    const int t  = threadIdx.x;
    const int r  = t >> 3;                     // 0..31
    const int c8 = (t & 7) * 8;                // 0,8,..,56
    #pragma unroll
    for (int h = 0; h < 2; ++h) {
        int kk = r + h * 32;
        v8h v = *(const v8h*)(Xh + (size_t)(k0 + kk) * DIM + d0 + c8);
        #pragma unroll
        for (int i = 0; i < 8; ++i) tile[kk][c8 + i] = v[i];
    }
    __syncthreads();
    #pragma unroll
    for (int h = 0; h < 2; ++h) {
        int dd = r + h * 32;
        v8h v;
        #pragma unroll
        for (int i = 0; i < 8; ++i) v[i] = tile[c8 + i][dd];
        *(v8h*)(Xth + (size_t)(d0 + dd) * K_TOK + k0 + c8) = v;
    }
}

// ---------------------------------------------------------------------------
// dots: dot[k] = p . x_k via WMMA with p BROADCAST into all 16 B-columns
// (uniform loads, no EXEC masking). Then per-token log_map_at scalar math ->
// wh[k] (coeff on x_k) and sA (coeff on p). 8 waves/block, 1 tile/wave.
__global__ void dots_weights_kernel(const _Float16* __restrict__ Xh,
                                    const _Float16* __restrict__ ph,
                                    const float* __restrict__ x_sq,
                                    const float* __restrict__ p_sq_ptr,
                                    float* __restrict__ dot, _Float16* __restrict__ wh,
                                    float* __restrict__ sA_slot) {
    const int lane = threadIdx.x & 31;
    const int wave = threadIdx.x >> 5;
    const int tile = blockIdx.x * 8 + wave;
    const int k0   = tile * 16;
    const int m    = lane & 15;
    const int hi   = lane >> 4;

    v8f c = {};
    const _Float16* arow = Xh + (size_t)(k0 + m) * DIM;
    for (int d0 = 0; d0 < DIM; d0 += 32) {
        // A-frag (16x32 f16): lane m, K-groups {d0+hi*8..+7} and {+16..+23}
        int ab = d0 + hi * 8;
        v8h lo = *(const v8h*)(arow + ab);
        v8h h8 = *(const v8h*)(arow + ab + 16);
        v16h a;
        #pragma unroll
        for (int i = 0; i < 8; ++i) { a[i] = lo[i]; a[8 + i] = h8[i]; }
        // B-frag (32x16): p chunk broadcast to every column -> uniform load
        v8h blo = *(const v8h*)(ph + ab);
        v8h bhi = *(const v8h*)(ph + ab + 16);
        v16h b;
        #pragma unroll
        for (int i = 0; i < 8; ++i) { b[i] = blo[i]; b[8 + i] = bhi[i]; }
        c = __builtin_amdgcn_wmma_f32_16x16x32_f16(false, a, false, b, (short)0, c,
                                                   false, false);
    }
    // Every column of C holds the dots: lane 0 -> rows k0..k0+7 (c[0..7]),
    // lane 16 -> rows k0+8..k0+15.
    if (m == 0) {
        float psq   = *p_sq_ptr;
        float psq_c = fminf(psq, 0.99f);
        float beta  = 1.0f - psq_c;
        float sA_loc = 0.0f;
        #pragma unroll
        for (int r = 0; r < 8; ++r) {
            int   row = k0 + hi * 8 + r;
            float dv  = c[r];
            dot[row]  = dv;
            float xs   = x_sq[row];
            float xs_c = fminf(xs, 0.99f);
            float alpha = 1.0f - 2.0f * dv + xs_c;
            float den   = fmaxf(1.0f - 2.0f * dv + psq_c * xs_c, EPSV);
            float nsq = alpha * alpha * psq - 2.0f * alpha * beta * dv + beta * beta * xs;
            nsq = fmaxf(nsq, 0.0f);
            float rn = sqrtf(nsq) / den;
            float cf = (rn >= MAXN) ? (MAXN - EPSV) / rn : 1.0f;
            float n  = rn * cf;
            float wv = 0.0f, sv = 0.0f;
            if (n >= EPSV) {
                float s  = fminf(n, MAXN);
                float at = 0.5f * logf((1.0f + s) / (1.0f - s));
                float sc = 2.0f * at / (n + EPSV);
                float f  = sc * cf / den;
                wv =  f * beta;    // coefficient on x_k
                sv = -f * alpha;   // coefficient on p
            }
            wh[row] = (_Float16)wv;
            sA_loc += sv;
        }
        atomicAdd(sA_slot, sA_loc);
    }
}

// ---------------------------------------------------------------------------
// g[d] = (X^T w)[d] + sA * p[d] via WMMA with w BROADCAST into all 16 A-rows
// (uniform loads). Also reduces |g|^2 and p.g into per-iteration slots.
__global__ void grad_kernel(const _Float16* __restrict__ Xth,
                            const _Float16* __restrict__ wh,
                            const float* __restrict__ p,
                            const float* __restrict__ sA_slot,
                            float* __restrict__ g,
                            float* __restrict__ gg_slot, float* __restrict__ pg_slot) {
    const int lane = threadIdx.x & 31;
    const int wave = threadIdx.x >> 5;
    const int tile = blockIdx.x * 8 + wave;
    const int d0   = tile * 16;
    const int m    = lane & 15;
    const int hi   = lane >> 4;

    v8f c = {};
    const _Float16* brow = Xth + (size_t)(d0 + m) * K_TOK;
    for (int k0 = 0; k0 < K_TOK; k0 += 32) {
        // A-frag: w chunk broadcast to every row -> uniform load
        int ab = k0 + hi * 8;
        v8h lo = *(const v8h*)(wh + ab);
        v8h h8 = *(const v8h*)(wh + ab + 16);
        v16h a;
        #pragma unroll
        for (int i = 0; i < 8; ++i) { a[i] = lo[i]; a[8 + i] = h8[i]; }
        // B-frag: lane n = column d0+n; K groups {k0+hi*8..+7} and {+16..+23}
        v8h blo = *(const v8h*)(brow + ab);
        v8h bhi = *(const v8h*)(brow + ab + 16);
        v16h b;
        #pragma unroll
        for (int i = 0; i < 8; ++i) { b[i] = blo[i]; b[8 + i] = bhi[i]; }
        c = __builtin_amdgcn_wmma_f32_16x16x32_f16(false, a, false, b, (short)0, c,
                                                   false, false);
    }
    float gv = 0.0f, pv = 0.0f;
    if (lane < 16) {                 // every row of C identical; take M=0: c[0]
        int d = d0 + lane;
        float sA = *sA_slot;
        pv = p[d];
        gv = c[0] + sA * pv;
        g[d] = gv;
    }
    float ggp = wave_sum32(gv * gv);
    float pgp = wave_sum32(gv * pv);
    if (lane == 0) { atomicAdd(gg_slot, ggp); atomicAdd(pg_slot, pgp); }
}

// ---------------------------------------------------------------------------
// center0 = exp_map0(v0), v0 == g buffer, |v0|^2 == gg_slot.  One block.
__global__ void init_center_kernel(const float* __restrict__ g,
                                   const float* __restrict__ gg_slot,
                                   float* __restrict__ p, _Float16* __restrict__ ph,
                                   float* __restrict__ p_sq) {
    float gg = *gg_slot;
    float n  = sqrtf(gg);
    float sc = (n < EPSV) ? 0.0f : tanhf(n) / (n + EPSV);
    for (int j = threadIdx.x; j < DIM; j += 256) {
        float v = sc * g[j];
        p[j] = v; ph[j] = (_Float16)v;
    }
    if (threadIdx.x == 0) *p_sq = sc * sc * gg;
}

// ---------------------------------------------------------------------------
// center = exp_map_at(p, -step*g): fully scalar coefficients + one AXPY.
__global__ void update_center_kernel(const float* __restrict__ g,
                                     const float* __restrict__ gg_slot,
                                     const float* __restrict__ pg_slot,
                                     float* __restrict__ p, _Float16* __restrict__ ph,
                                     float* __restrict__ p_sq) {
    float gg  = *gg_slot;
    float pg  = *pg_slot;
    float psq = *p_sq;
    const float step = 0.1f / (float)K_TOK;
    float nu = sqrtf(step * step * gg);
    if (nu < EPSV) return;                         // uniform: center unchanged
    float sv  = tanhf(0.5f * nu) / (nu + EPSV);
    float t   = -step * sv;                        // q = t * g
    float qsq = t * t * gg;
    float xy  = t * pg;
    float xs = fminf(psq, 0.99f), ys = fminf(qsq, 0.99f);
    float cA  = 1.0f + 2.0f * xy + ys;
    float cB  = 1.0f - xs;
    float den = fmaxf(1.0f + 2.0f * xy + xs * ys, EPSV);
    float nsq = fmaxf(cA * cA * psq + 2.0f * cA * cB * xy + cB * cB * qsq, 0.0f);
    float rn  = sqrtf(nsq) / den;
    float cf  = (rn >= MAXN) ? (MAXN - EPSV) / rn : 1.0f;
    float fa = cf * cA / den, fb = cf * cB * t / den;
    for (int j = threadIdx.x; j < DIM; j += 256) {
        float v = fa * p[j] + fb * g[j];
        p[j] = v; ph[j] = (_Float16)v;
    }
    __syncthreads();
    if (threadIdx.x == 0) *p_sq = cf * cf * nsq / (den * den);
}

// ---------------------------------------------------------------------------
// loss = mean_k max(asinh(sqrt(max(|x-p|^2/den,0))) - R, 0)
__global__ void loss_kernel(const float* __restrict__ x_sq, const float* __restrict__ dot,
                            const float* __restrict__ p_sq, float* __restrict__ out) {
    int k = blockIdx.x * blockDim.x + threadIdx.x;
    float psq = *p_sq;
    float viol = 0.0f;
    if (k < K_TOK) {
        float xs  = x_sq[k];
        float dv  = dot[k];
        float dsq = fmaxf(xs + psq - 2.0f * dv, 0.0f);
        float den = fmaxf((1.0f - xs) * (1.0f - psq), EPSV);
        float s   = sqrtf(fmaxf(dsq / den, 0.0f));
        viol = fmaxf(asinhf(s) - RADIUS_T, 0.0f);
    }
    __shared__ float red[256];
    red[threadIdx.x] = viol; __syncthreads();
    for (int s = 128; s > 0; s >>= 1) {
        if (threadIdx.x < s) red[threadIdx.x] += red[threadIdx.x + s];
        __syncthreads();
    }
    if (threadIdx.x == 0) atomicAdd(out, red[0] * (1.0f / (float)K_TOK));
}

// ---------------------------------------------------------------------------
extern "C" void kernel_launch(void* const* d_in, const int* in_sizes, int n_in,
                              void* d_out, int out_size, void* d_ws, size_t ws_size,
                              hipStream_t stream) {
    const float* in = (const float*)d_in[0];
    float* out = (float*)d_out;

    // workspace layout (f16 blocks first for 32B alignment of fragment loads)
    char* w = (char*)d_ws;
    _Float16* Xh  = (_Float16*)w;                          // K*D f16   (8 MB)
    _Float16* Xth = Xh + (size_t)K_TOK * DIM;              // D*K f16   (8 MB)
    _Float16* ph  = Xth + (size_t)DIM * K_TOK;             // D f16
    _Float16* wh  = ph + DIM;                              // K f16
    _Float16* w0h = wh + K_TOK;                            // K f16
    float* x_sq = (float*)(w0h + K_TOK);                   // K f32
    float* dot  = x_sq + K_TOK;                            // K f32
    float* g    = dot + K_TOK;                             // D f32
    float* p    = g + DIM;                                 // D f32
    float* acc  = p + DIM;                                 // NACC f32 slots
    float* p_sq = acc + NACC;                              // 1 f32
    float* sA = acc;            // slots 0..49 (slot 63 stays zero for init)
    float* gg = acc + 64;       // slots 0..49, slot 50 = |v0|^2
    float* pg = acc + 128;      // slots 0..49, slot 50 unused

    zero_state_kernel<<<8, 256, 0, stream>>>(p, acc, out);
    tokens_kernel<<<K_TOK, 256, 0, stream>>>(in, Xh, x_sq, w0h);
    transpose_kernel<<<1024, 256, 0, stream>>>(Xh, Xth);

    // Karcher init: v0 = X^T w0 (sA slot 63 == 0, p == 0), then exp_map0
    grad_kernel<<<16, 256, 0, stream>>>(Xth, w0h, p, &sA[63], g, &gg[50], &pg[50]);
    init_center_kernel<<<1, 256, 0, stream>>>(g, &gg[50], p, ph, p_sq);

    for (int it = 0; it < NITER; ++it) {
        dots_weights_kernel<<<16, 256, 0, stream>>>(Xh, ph, x_sq, p_sq, dot, wh, &sA[it]);
        grad_kernel<<<16, 256, 0, stream>>>(Xth, wh, p, &sA[it], g, &gg[it], &pg[it]);
        update_center_kernel<<<1, 256, 0, stream>>>(g, &gg[it], &pg[it], p, ph, p_sq);
    }

    // final dots against converged center (sA slot 62 is a write-only dump)
    dots_weights_kernel<<<16, 256, 0, stream>>>(Xh, ph, x_sq, p_sq, dot, wh, &sA[62]);
    loss_kernel<<<8, 256, 0, stream>>>(x_sq, dot, p_sq, out);
}
